// SelfAttention_23897198035647
// MI455X (gfx1250) — matrix operand order repaired and checked by reading, not compiled, not run on hardware
//
#include <hip/hip_runtime.h>

// ---------------------------------------------------------------------------
// Self-attention without softmax:  out = (x@Wq) @ ((x@Wk)^T (x@Wv))  per head.
// Reassociation: (q k^T) v == q (k^T v); k^T v is only 64x64 per (b,h).
// All matrix math on v_wmma_f32_16x16x32_bf16 (bf16 in, f32 accum).
// K/V tile staging uses global_load_async_to_lds_b128 (ASYNCcnt).
// B=8, N=2048, C=512, H=8, D=64.  qkv row-major (16384 x 1536) bf16 in ws.
// ---------------------------------------------------------------------------

typedef __attribute__((ext_vector_type(16))) __bf16 v16bf;
typedef __attribute__((ext_vector_type(8)))  float  v8f;
typedef __attribute__((ext_vector_type(4)))  int    v4i;

typedef __attribute__((address_space(1))) v4i* gv4i_p;   // global b128 ptr
typedef __attribute__((address_space(3))) v4i* lv4i_p;   // LDS b128 ptr

#if defined(__has_builtin)
#if __has_builtin(__builtin_amdgcn_global_load_async_to_lds_b128)
#define HAVE_ASYNC_LDS 1
#endif
#endif

union Frag {
    v16bf v;
    unsigned u[8];
};

__device__ __forceinline__ unsigned short f32_to_bf16(float f) {
    unsigned u = __float_as_uint(f);
    unsigned rnd = 0x7FFFu + ((u >> 16) & 1u);   // round-to-nearest-even
    return (unsigned short)((u + rnd) >> 16);
}

// A-matrix (16x32 bf16) K-pair start for VGPR j, per ISA 7.12.2:
//   lanes 0-15:  V0..3 -> K=0..7,  V4..7 -> K=16..23
//   lanes 16-31: V0..3 -> K=8..15, V4..7 -> K=24..31
__device__ __forceinline__ int kpairA(int lane, int j) {
    int h = lane >> 4;
    return (j < 4) ? (8 * h + 2 * j) : (16 + 8 * h + 2 * (j - 4));
}

// B-matrix (32x16 bf16) K-pair start for VGPR j (same K order as A):
//   lanes 0-15:  V0..3 -> K=0..7,  V4..7 -> K=16..23
//   lanes 16-31: V0..3 -> K=8..15, V4..7 -> K=24..31
__device__ __forceinline__ int kpairB(int lane, int j) {
    int h = lane >> 4;
    return (j < 4) ? (8 * h + 2 * j) : (16 + 8 * h + 2 * (j - 4));
}

__device__ __forceinline__ v8f wmma_bf16(const Frag& a, const Frag& b, v8f c) {
    return __builtin_amdgcn_wmma_f32_16x16x32_bf16(
        false, a.v, false, b.v, (short)0, c, false, false);
}

// ---------------------------------------------------------------------------
// Kernel 1: qkv_bf16[16384,1536] = bf16(x[16384,512] @ w[512,1536])
// Block tile 128x128, K-step 32, 8 waves: wave (wm=w&3, wn=w>>2) owns a
// 32x64 sub-tile (2x4 WMMA tiles). LDS: As[m][k], Bt[n][k] (k contiguous).
// (Staging must run through VGPRs anyway: f32 -> bf16 conversion.)
// ---------------------------------------------------------------------------
__global__ __launch_bounds__(256) void gemm_qkv_kernel(
    const float* __restrict__ x, const float* __restrict__ w,
    unsigned short* __restrict__ qkv) {
    __shared__ __align__(16) unsigned short As[128][32];
    __shared__ __align__(16) unsigned short Bt[128][32];

    const int t    = threadIdx.x;
    const int lane = t & 31;
    const int wave = t >> 5;
    const int wm   = wave & 3;   // row group: wm*32
    const int wn   = wave >> 2;  // col group: wn*64
    const int m0   = blockIdx.y * 128;
    const int n0   = blockIdx.x * 128;

    v8f acc[2][4] = {};

    for (int k0 = 0; k0 < 512; k0 += 32) {
        // Stage A: 128x32 f32 -> bf16, coalesced along k.
        for (int i = 0; i < 16; ++i) {
            int idx = t + i * 256;
            int m = idx >> 5, kk = idx & 31;
            As[m][kk] = f32_to_bf16(x[(size_t)(m0 + m) * 512 + k0 + kk]);
        }
        // Stage B transposed: coalesced along n in global, [n][k] in LDS.
        for (int i = 0; i < 16; ++i) {
            int idx = t + i * 256;
            int kk = idx >> 7, n = idx & 127;
            Bt[n][kk] = f32_to_bf16(w[(size_t)(k0 + kk) * 1536 + n0 + n]);
        }
        if (k0 + 32 < 512)  // hint next A tile into cache (global_prefetch_b8)
            __builtin_prefetch(&x[(size_t)(m0 + (t >> 1)) * 512 + k0 + 32], 0, 1);
        __syncthreads();

        Frag a[2], b[4];
        for (int mt = 0; mt < 2; ++mt) {
            int row = wm * 32 + mt * 16 + (lane & 15);
            for (int j = 0; j < 8; ++j)
                a[mt].u[j] = *(const unsigned*)&As[row][kpairA(lane, j)];
        }
        for (int nt = 0; nt < 4; ++nt) {
            int col = wn * 64 + nt * 16 + (lane & 15);
            for (int j = 0; j < 8; ++j)
                b[nt].u[j] = *(const unsigned*)&Bt[col][kpairB(lane, j)];
        }
        for (int mt = 0; mt < 2; ++mt)
            for (int nt = 0; nt < 4; ++nt)
                acc[mt][nt] = wmma_bf16(a[mt], b[nt], acc[mt][nt]);
        __syncthreads();
    }

    // D layout: VGPR r -> row r + 8*(lane>=16), col lane&15.
    for (int mt = 0; mt < 2; ++mt)
        for (int nt = 0; nt < 4; ++nt)
            for (int r = 0; r < 8; ++r) {
                int grow = m0 + wm * 32 + mt * 16 + r + ((lane >> 4) << 3);
                int gcol = n0 + wn * 64 + nt * 16 + (lane & 15);
                qkv[(size_t)grow * 1536 + gcol] = f32_to_bf16(acc[mt][nt][r]);
            }
}

// ---------------------------------------------------------------------------
// Kernel 2: per (b,h):  St[d2][d1] = sum_n V[n][d2] * K[n][d1]   (= (k^T v)^T)
// One wave per block; streams 32-row K/V chunks through LDS via async-to-LDS
// copies (bf16 already, no conversion needed); 4x4 WMMA tiles cover the full
// 64x64; result stored transposed, k-contiguous, as bf16.
// ---------------------------------------------------------------------------
__global__ __launch_bounds__(32) void ktv_kernel(
    const unsigned short* __restrict__ qkv, unsigned short* __restrict__ St) {
    const int bh = blockIdx.x;          // b*8 + h
    const int b = bh >> 3, h = bh & 7;
    const int lane = threadIdx.x;

    __shared__ __align__(16) unsigned short kS[32][64];   // [n][d1]
    __shared__ __align__(16) unsigned short vS[32][64];   // [n][d2]

    v8f acc[4][4] = {};

    const unsigned short* kbase = qkv + (size_t)b * 2048 * 1536 + 512  + h * 64;
    const unsigned short* vbase = qkv + (size_t)b * 2048 * 1536 + 1024 + h * 64;

    for (int nc = 0; nc < 64; ++nc) {
        const int n = nc * 32 + lane;             // each lane stages one row
        const unsigned short* kg = kbase + (size_t)n * 1536;
        const unsigned short* vg = vbase + (size_t)n * 1536;
#if HAVE_ASYNC_LDS
        // 128 B per lane per matrix, 16 B granules; global & LDS addresses
        // advance in lock-step so per-lane address semantics match.
        unsigned short* kl = &kS[lane][0];
        unsigned short* vl = &vS[lane][0];
        for (int i = 0; i < 8; ++i) {
            __builtin_amdgcn_global_load_async_to_lds_b128(
                (gv4i_p)(kg + i * 8), (lv4i_p)(kl + i * 8), 0, 0);
            __builtin_amdgcn_global_load_async_to_lds_b128(
                (gv4i_p)(vg + i * 8), (lv4i_p)(vl + i * 8), 0, 0);
        }
#if defined(__has_builtin) && __has_builtin(__builtin_amdgcn_s_wait_asynccnt)
        __builtin_amdgcn_s_wait_asynccnt(0);
#else
        asm volatile("s_wait_asynccnt 0x0" ::: "memory");
#endif
#else
        {   // fallback: VGPR-relayed staging
            const uint4* krow = (const uint4*)kg;
            const uint4* vrow = (const uint4*)vg;
            uint4* kdst = (uint4*)&kS[lane][0];
            uint4* vdst = (uint4*)&vS[lane][0];
            for (int i = 0; i < 8; ++i) kdst[i] = krow[i];
            for (int i = 0; i < 8; ++i) vdst[i] = vrow[i];
        }
#endif
        __syncthreads();

        Frag a[4], bf[4];
        for (int dt = 0; dt < 4; ++dt) {
            int d = dt * 16 + (lane & 15);
            for (int j = 0; j < 8; ++j) {         // A[m][kk] = K[n0+kk][d1]
                int kk = kpairA(lane, j);
                unsigned lo = kS[kk][d], hi = kS[kk + 1][d];
                a[dt].u[j] = lo | (hi << 16);
            }
            for (int j = 0; j < 8; ++j) {         // B[kk][n] = V[n0+kk][d2]
                int kk = kpairB(lane, j);
                unsigned lo = vS[kk][d], hi = vS[kk + 1][d];
                bf[dt].u[j] = lo | (hi << 16);
            }
        }
        for (int i = 0; i < 4; ++i)
            for (int j = 0; j < 4; ++j)
                acc[i][j] = wmma_bf16(a[i], bf[j], acc[i][j]);
        __syncthreads();
    }

    unsigned short* stp = St + (size_t)bh * 4096;
    for (int i = 0; i < 4; ++i)
        for (int j = 0; j < 4; ++j)
            for (int r = 0; r < 8; ++r) {
                int d1 = i * 16 + r + ((lane >> 4) << 3);
                int d2 = j * 16 + (lane & 15);
                stp[d2 * 64 + d1] = f32_to_bf16(acc[i][j][r]);  // transposed
            }
}

// ---------------------------------------------------------------------------
// Kernel 3: out[b, n, h*64+d2] = sum_d1 q[b,h,n,d1] * St[d2][d1]   (f32 out)
// One wave per (b,h, 16-row tile). K=64 -> 2 WMMA chunks. Fragments are
// contiguous dword loads straight from global (q is k-contiguous; St stored
// d2-major k-contiguous by kernel 2).
// ---------------------------------------------------------------------------
__global__ __launch_bounds__(32) void qs_kernel(
    const unsigned short* __restrict__ qkv,
    const unsigned short* __restrict__ St, float* __restrict__ out) {
    const int bh = blockIdx.y;
    const int b = bh >> 3, h = bh & 7;
    const int n0 = blockIdx.x * 16;
    const int lane = threadIdx.x;

    v8f acc[4] = {};
    const unsigned short* qb = qkv + (size_t)b * 2048 * 1536 + h * 64;
    const unsigned short* sp = St + (size_t)bh * 4096;

    for (int c = 0; c < 2; ++c) {
        Frag a;
        const unsigned short* qrow =
            qb + (size_t)(n0 + (lane & 15)) * 1536 + c * 32;
        for (int j = 0; j < 8; ++j)
            a.u[j] = *(const unsigned*)(qrow + kpairA(lane, j));
        for (int nt = 0; nt < 4; ++nt) {
            Frag bf;
            const unsigned short* srow =
                sp + (nt * 16 + (lane & 15)) * 64 + c * 32;
            for (int j = 0; j < 8; ++j)
                bf.u[j] = *(const unsigned*)(srow + kpairB(lane, j));
            acc[nt] = wmma_bf16(a, bf, acc[nt]);
        }
    }

    float* ob = out + (size_t)b * 2048 * 512;
    for (int nt = 0; nt < 4; ++nt)
        for (int r = 0; r < 8; ++r) {
            int n  = n0 + r + ((lane >> 4) << 3);
            int cg = h * 64 + nt * 16 + (lane & 15);
            ob[(size_t)n * 512 + cg] = acc[nt][r];
        }
}

// ---------------------------------------------------------------------------
extern "C" void kernel_launch(void* const* d_in, const int* in_sizes, int n_in,
                              void* d_out, int out_size, void* d_ws, size_t ws_size,
                              hipStream_t stream) {
    (void)in_sizes; (void)n_in; (void)out_size; (void)ws_size;
    const float* x = (const float*)d_in[0];        // (8,2048,512) f32
    const float* w = (const float*)d_in[1];        // (512,1536)  f32
    float* out = (float*)d_out;                    // (8,2048,512) f32

    unsigned short* qkv = (unsigned short*)d_ws;               // 16384x1536 bf16 (~50.3 MB)
    unsigned short* St  = qkv + (size_t)16384 * 1536;          // 64x(64x64)  bf16 (0.5 MB)

    dim3 g1(12, 128);                // N/128, M/128
    gemm_qkv_kernel<<<g1, 256, 0, stream>>>(x, w, qkv);

    ktv_kernel<<<64, 32, 0, stream>>>(qkv, St);

    dim3 g3(128, 64);                // n-tiles, b*h
    qs_kernel<<<g3, 32, 0, stream>>>(qkv, St, out);
}
